// ManualRNN_1958505087665
// MI455X (gfx1250) — matrix operand order repaired and checked by reading, not compiled, not run on hardware
//
#include <hip/hip_runtime.h>
#include <math.h>

typedef __attribute__((ext_vector_type(16))) _Float16 v16h;
typedef __attribute__((ext_vector_type(8)))  _Float16 v8h;
typedef __attribute__((ext_vector_type(8)))  float    v8f;

#define B_  256
#define T_  512
#define I_  512
#define H_  1024
#define O_  512
#define KC_ (I_ + H_)   // 1536 combined K for [x_t | h] @ [Wx ; Wh]
#define NWG 32          // 4 (M) x 8 (N) workgroups, each 64x128 of h[256x1024]
#define KCH 64          // K chunk staged in LDS
#define LDA 72          // padded LDS row stride (halves): 64+8 -> bank-conflict-free frags
#define LDB 72

// ---------- WMMA ----------
__device__ __forceinline__ v8f wmma16(v16h a, v16h b, v8f c) {
  return __builtin_amdgcn_wmma_f32_16x16x32_f16(
      false, a, false, b, (short)0, c, false, false);
}

union V16 { v16h v; v8h h[2]; };

// A frag: lane l -> row M, K = {kb..kb+7, kb+16..kb+23}, kb = base + 8*(l>=16)
__device__ __forceinline__ v16h ldsFragA(const _Float16* row, int k) {
  V16 u;
  u.h[0] = *(const v8h*)(row + k);
  u.h[1] = *(const v8h*)(row + k + 16);
  return u.v;
}
// B frag: lane l -> col N=(l&15), K = {kb..kb+7, kb+16..kb+23}, kb = base + 8*(l>=16)
__device__ __forceinline__ v16h ldsFragB(const _Float16* row, int k) {
  V16 u;
  u.h[0] = *(const v8h*)(row + k);
  u.h[1] = *(const v8h*)(row + k + 16);
  return u.v;
}

__device__ __forceinline__ float fast_tanh(float v) {
#if __has_builtin(__builtin_amdgcn_tanhf)
  return __builtin_amdgcn_tanhf(v);
#elif __has_builtin(__builtin_amdgcn_tanh_f32)
  return __builtin_amdgcn_tanh_f32(v);
#else
  return tanhf(v);
#endif
}

// ---------- CDNA5 async global->LDS copy (ASYNCcnt path) ----------
// Copies 16B per lane from global memory directly into LDS, no VGPR data.
__device__ __forceinline__ void asyncCopy16(unsigned ldsOff, const void* gsrc) {
  asm volatile("global_load_async_to_lds_b128 %0, %1, off"
               :: "v"(ldsOff), "v"(gsrc) : "memory");
}
__device__ __forceinline__ void asyncFence() {
  asm volatile("s_wait_asynccnt 0x0" ::: "memory");
}
__device__ __forceinline__ unsigned ldsAddr(const void* p) {
  return (unsigned)(size_t)p;   // low 32 bits of generic LDS pointer = LDS offset
}

// ---------- device-wide split barrier ----------
__device__ __forceinline__ unsigned gridArrive(unsigned* cnt, unsigned* gen) {
  __threadfence();
  __syncthreads();
  unsigned g = 0;
  if (threadIdx.x == 0) {
    g = __hip_atomic_load(gen, __ATOMIC_ACQUIRE, __HIP_MEMORY_SCOPE_AGENT);
    unsigned a = __hip_atomic_fetch_add(cnt, 1u, __ATOMIC_ACQ_REL, __HIP_MEMORY_SCOPE_AGENT);
    if (a == NWG - 1) {
      __hip_atomic_store(cnt, 0u, __ATOMIC_RELAXED, __HIP_MEMORY_SCOPE_AGENT);
      __hip_atomic_fetch_add(gen, 1u, __ATOMIC_RELEASE, __HIP_MEMORY_SCOPE_AGENT);
    }
  }
  return g;
}
__device__ __forceinline__ void gridWait(unsigned* gen, unsigned g) {
  if (threadIdx.x == 0) {
    while (__hip_atomic_load(gen, __ATOMIC_ACQUIRE, __HIP_MEMORY_SCOPE_AGENT) == g)
      __builtin_amdgcn_s_sleep(2);
  }
  __syncthreads();
}

// ---------- prep: transpose + convert weights to f16 ----------
__global__ void rnn_prep(const float* __restrict__ Wx, const float* __restrict__ Wh,
                         const float* __restrict__ Wy,
                         _Float16* __restrict__ WcT, _Float16* __restrict__ WyT) {
  int i = blockIdx.x * blockDim.x + threadIdx.x;
  const int totalC = H_ * KC_;
  if (i < totalC) {
    int n = i / KC_, k = i % KC_;
    float v = (k < I_) ? Wx[(size_t)k * H_ + n] : Wh[(size_t)(k - I_) * H_ + n];
    WcT[i] = (_Float16)v;
  } else {
    int j = i - totalC;
    if (j < O_ * H_) {
      int o = j / H_, k = j % H_;
      WyT[j] = (_Float16)Wy[(size_t)k * O_ + o];
    }
  }
}

struct Acc4 { v8f c00, c01, c10, c11; };

// ---------- persistent RNN kernel ----------
__global__ __launch_bounds__(256, 1)
void rnn_persist(const float* __restrict__ x,      // [B,T,I]
                 const float* __restrict__ bh,     // [H]
                 const float* __restrict__ by,     // [O]
                 const _Float16* __restrict__ WcT, // [H][KC]
                 const _Float16* __restrict__ WyT, // [O][H]
                 _Float16* __restrict__ h0,        // [B][H] (init 0)
                 _Float16* __restrict__ h1,        // [B][H]
                 float* __restrict__ y,            // [B][O]
                 unsigned* __restrict__ syncCnt, unsigned* __restrict__ syncGen) {
  __shared__ _Float16 As[2][64][LDA];    // A tile: 64 rows x 64 K (padded)
  __shared__ _Float16 Bs[2][128][LDB];   // B tile: 128 cols x 64 K (padded)

  const int tid  = threadIdx.x;
  const int wave = tid >> 5;
  const int lane = tid & 31;
  const int waveM = wave & 1;          // 2 waves in M
  const int waveN = wave >> 1;         // 4 waves in N
  const int l15  = lane & 15;
  const int half = lane >> 4;
  const int wgM  = blockIdx.x >> 3;    // 0..3
  const int wgN  = blockIdx.x & 7;     // 0..7
  const int rowBase = wgM * 64;
  const int colBase = wgN * 128;
  const int mW = rowBase + waveM * 32; // wave owns rows mW..mW+31 (2 tiles)
  const int nW = colBase + waveN * 32; // wave owns cols nW..nW+31 (2 tiles)
  const int rowOff = half * 8;

  // cooperative fill indices
  const int fr = tid >> 2;             // A fill: row 0..63
  const int fk = (tid & 3) << 4;       // A fill: k 0/16/32/48 (16 halves each)
  const int bn = tid >> 1;             // B fill: col 0..127
  const int bk = (tid & 1) << 5;       // B fill: k 0/32 (32 halves each)

  // wave-local LDS fragment coordinates
  const int aR0 = waveM * 32 + l15;
  const int aR1 = aR0 + 16;
  const int bC0 = waveN * 32 + l15;
  const int bC1 = bC0 + 16;
  const int aKo = half * 8;
  const int bKo = half * 8;

  const float bh0 = bh[nW + l15];
  const float bh1 = bh[nW + 16 + l15];

  // f16->f16 tile fills go straight memory->LDS on the async engine.
  auto fillA_h = [&](int buf, const _Float16* hsrc, int kc) {
    const _Float16* s = hsrc + (size_t)(rowBase + fr) * H_ + kc + fk;
    unsigned d = ldsAddr(&As[buf][fr][fk]);
    asyncCopy16(d,      s);
    asyncCopy16(d + 16, s + 8);
  };
  auto fillB = [&](int buf, const _Float16* WT, int ld, int kc) {
    const _Float16* s = WT + (size_t)(colBase + bn) * ld + kc + bk;
    unsigned d = ldsAddr(&Bs[buf][bn][bk]);
    asyncCopy16(d,      s);
    asyncCopy16(d + 16, s + 8);
    asyncCopy16(d + 32, s + 16);
    asyncCopy16(d + 48, s + 24);
  };
  // x is fp32 in memory: load, convert to f16 in registers, store to LDS.
  auto fillA_x = [&](int buf, int t, int kc) {
    const float* s = x + (size_t)(rowBase + fr) * (T_ * I_) + (size_t)t * I_ + kc + fk;
    v8f f0 = *(const v8f*)s;
    v8f f1 = *(const v8f*)(s + 8);
    v8h p0, p1;
#pragma unroll
    for (int i = 0; i < 8; ++i) { p0[i] = (_Float16)f0[i]; p1[i] = (_Float16)f1[i]; }
    *(v8h*)&As[buf][fr][fk]     = p0;
    *(v8h*)&As[buf][fr][fk + 8] = p1;
  };
  // tile handoff: drain async engine, then workgroup barrier
  auto syncTile = [&]() {
    asyncFence();
    __syncthreads();
  };
  auto compute = [&](int buf, Acc4& C) {
#pragma unroll
    for (int kk = 0; kk < KCH; kk += 32) {
      v16h a0 = ldsFragA(&As[buf][aR0][0], kk + aKo);
      v16h a1 = ldsFragA(&As[buf][aR1][0], kk + aKo);
      v16h b0 = ldsFragB(&Bs[buf][bC0][0], kk + bKo);
      v16h b1 = ldsFragB(&Bs[buf][bC1][0], kk + bKo);
      C.c00 = wmma16(a0, b0, C.c00);
      C.c01 = wmma16(a0, b1, C.c01);
      C.c10 = wmma16(a1, b0, C.c10);
      C.c11 = wmma16(a1, b1, C.c11);
    }
  };

  // x_t @ Wx : K = 0..511 against WcT[:, 0:512]; independent of h -> used to
  // hide the device-wide barrier latency.
  auto xpart = [&](int t) -> Acc4 {
    Acc4 R; R.c00 = v8f{}; R.c01 = v8f{}; R.c10 = v8f{}; R.c11 = v8f{};
    const int NCH = I_ / KCH;  // 8 chunks
    fillA_x(0, t, 0);
    fillB(0, WcT, KC_, 0);
    syncTile();
    for (int i = 0; i < NCH; ++i) {
      int cur = i & 1;
      if (i + 1 < NCH) {
        fillA_x(cur ^ 1, t, (i + 1) * KCH);
        fillB(cur ^ 1, WcT, KC_, (i + 1) * KCH);
      }
      compute(cur, R);
      syncTile();
    }
    return R;
  };

  Acc4 AX = xpart(0);
  unsigned g = 0;

  for (int t = 0; t < T_; ++t) {
    const _Float16* hc = (t & 1) ? h1 : h0;
    _Float16*       hn = (t & 1) ? h0 : h1;

    Acc4 C = AX;
    // h @ Wh : K = 512..1535 against WcT[:, 512:1536], 16 chunks, double-buffered
    {
      const int NCH = H_ / KCH;
      fillA_h(0, hc, 0);
      fillB(0, WcT, KC_, I_);
      syncTile();
      for (int i = 0; i < NCH; ++i) {
        int cur = i & 1;
        if (i + 1 < NCH) {
          fillA_h(cur ^ 1, hc, (i + 1) * KCH);
          fillB(cur ^ 1, WcT, KC_, I_ + (i + 1) * KCH);
        }
        compute(cur, C);
        syncTile();
      }
    }

    // epilogue: h_next = tanh(acc + bh)
#pragma unroll
    for (int v = 0; v < 8; ++v) {
      int m0 = mW + rowOff + v;
      int m1 = m0 + 16;
      hn[(size_t)m0 * H_ + nW + l15]      = (_Float16)fast_tanh(C.c00[v] + bh0);
      hn[(size_t)m0 * H_ + nW + 16 + l15] = (_Float16)fast_tanh(C.c01[v] + bh1);
      hn[(size_t)m1 * H_ + nW + l15]      = (_Float16)fast_tanh(C.c10[v] + bh0);
      hn[(size_t)m1 * H_ + nW + 16 + l15] = (_Float16)fast_tanh(C.c11[v] + bh1);
    }

    g = gridArrive(syncCnt, syncGen);
    if (t + 1 < T_) AX = xpart(t + 1);  // overlap barrier wait with x-projection
    gridWait(syncGen, g);
  }

  // final: y = h_final @ Wy + by ; h_final in h0 (T even). WGs with wgN<4 cover
  // y[256x512] with 64x128 tiles.
  if (wgN < 4) {
    Acc4 C; C.c00 = v8f{}; C.c01 = v8f{}; C.c10 = v8f{}; C.c11 = v8f{};
    const int NCH = H_ / KCH;
    fillA_h(0, h0, 0);
    fillB(0, WyT, H_, 0);
    syncTile();
    for (int i = 0; i < NCH; ++i) {
      int cur = i & 1;
      if (i + 1 < NCH) {
        fillA_h(cur ^ 1, h0, (i + 1) * KCH);
        fillB(cur ^ 1, WyT, H_, (i + 1) * KCH);
      }
      compute(cur, C);
      syncTile();
    }
    const float by0 = by[nW + l15];
    const float by1 = by[nW + 16 + l15];
#pragma unroll
    for (int v = 0; v < 8; ++v) {
      int m0 = mW + rowOff + v;
      int m1 = m0 + 16;
      y[(size_t)m0 * O_ + nW + l15]      = C.c00[v] + by0;
      y[(size_t)m0 * O_ + nW + 16 + l15] = C.c01[v] + by1;
      y[(size_t)m1 * O_ + nW + l15]      = C.c10[v] + by0;
      y[(size_t)m1 * O_ + nW + 16 + l15] = C.c11[v] + by1;
    }
  }
}

// ---------- host launch ----------
extern "C" void kernel_launch(void* const* d_in, const int* in_sizes, int n_in,
                              void* d_out, int out_size, void* d_ws, size_t ws_size,
                              hipStream_t stream) {
  const float* x  = (const float*)d_in[0];
  const float* Wx = (const float*)d_in[1];
  const float* Wh = (const float*)d_in[2];
  const float* Wy = (const float*)d_in[3];
  const float* bh = (const float*)d_in[4];
  const float* by = (const float*)d_in[5];
  float* yOut = (float*)d_out;

  char* ws = (char*)d_ws;
  size_t oWcT  = 0;
  size_t oWyT  = oWcT + (size_t)H_ * KC_ * sizeof(_Float16);   // 3 MiB
  size_t oH0   = oWyT + (size_t)O_ * H_  * sizeof(_Float16);   // +1 MiB
  size_t oH1   = oH0  + (size_t)B_ * H_  * sizeof(_Float16);   // +0.5 MiB
  size_t oSync = oH1  + (size_t)B_ * H_  * sizeof(_Float16);   // +0.5 MiB

  _Float16* WcT = (_Float16*)(ws + oWcT);
  _Float16* WyT = (_Float16*)(ws + oWyT);
  _Float16* h0  = (_Float16*)(ws + oH0);
  _Float16* h1  = (_Float16*)(ws + oH1);
  unsigned* syncCnt = (unsigned*)(ws + oSync);
  unsigned* syncGen = syncCnt + 1;

  hipMemsetAsync(h0, 0, (size_t)B_ * H_ * sizeof(_Float16), stream);
  hipMemsetAsync(syncCnt, 0, 2 * sizeof(unsigned), stream);

  {
    int total = H_ * KC_ + O_ * H_;
    int blocks = (total + 255) / 256;
    rnn_prep<<<blocks, 256, 0, stream>>>(Wx, Wh, Wy, WcT, WyT);
  }

  rnn_persist<<<NWG, 256, 0, stream>>>(x, bh, by, WcT, WyT, h0, h1, yOut,
                                       syncCnt, syncGen);
}
